// MultiHeadAttention_32968168964630
// MI455X (gfx1250) — matrix operand order repaired and checked by reading, not compiled, not run on hardware
//
#include <hip/hip_runtime.h>
#include <hip/hip_bf16.h>

typedef __attribute__((ext_vector_type(16))) __bf16 bf16x16;
typedef __attribute__((ext_vector_type(8)))  float  f32x8;
typedef __attribute__((ext_vector_type(4)))  int    i32x4;

#define H_     16
#define L_     2048
#define B_     4
#define DHEAD  64
#define DMODEL 1024

typedef __attribute__((address_space(1))) i32x4 as1_i32x4;
typedef __attribute__((address_space(3))) i32x4 as3_i32x4;

// ---------------------------------------------------------------------------
// CDNA5 async copy: GLOBAL_LOAD_ASYNC_TO_LDS_B128 (ASYNCcnt-tracked).
// clang signature: (global int4*, lds int4*, imm offset, imm cpol)
// ---------------------------------------------------------------------------
__device__ __forceinline__ void async_copy16(const void* gsrc, void* ldst) {
#if __has_builtin(__builtin_amdgcn_global_load_async_to_lds_b128)
  __builtin_amdgcn_global_load_async_to_lds_b128(
      (as1_i32x4*)gsrc, (as3_i32x4*)ldst, 0, 0);
#else
  *(uint4*)ldst = *(const uint4*)gsrc;
#endif
}

template <int N>
__device__ __forceinline__ void wait_async() {
#if __has_builtin(__builtin_amdgcn_s_wait_asynccnt)
  __builtin_amdgcn_s_wait_asynccnt(N);
#else
  asm volatile("s_wait_asynccnt %0" ::"i"(N) : "memory");
#endif
}

// ---------------------------------------------------------------------------
// WMMA fragment helpers (CDNA5 wave32 layouts, CDNA5 ISA §7.12.2)
// ---------------------------------------------------------------------------
// A-matrix 16x32 bf16 fragment from a row-major tile.
__device__ __forceinline__ bf16x16 frag_a(const __bf16* base, int stride) {
  const int lane = threadIdx.x & 31;
  const int row  = lane & 15;
  const int kb   = (lane >> 4) << 3;     // 0 or 8
  const __bf16* p = base + row * stride + kb;
  bf16x16 f;
#pragma unroll
  for (int e = 0; e < 8; ++e) { f[e] = p[e]; f[e + 8] = p[e + 16]; }
  return f;
}

// B-matrix 32x16 bf16 fragment, sourced from B^T stored row-major.
__device__ __forceinline__ bf16x16 frag_bT(const __bf16* baseT, int stride) {
  const int lane = threadIdx.x & 31;
  const int col  = lane & 15;
  const int kb   = (lane >> 4) << 3;     // 0 or 8
  const __bf16* p = baseT + col * stride + kb;
  bf16x16 f;
#pragma unroll
  for (int e = 0; e < 8; ++e) { f[e] = p[e]; f[e + 8] = p[e + 16]; }
  return f;
}

__device__ __forceinline__ f32x8 wmma_bf16(bf16x16 a, bf16x16 b, f32x8 c) {
  return __builtin_amdgcn_wmma_f32_16x16x32_bf16(false, a, false, b,
                                                 (short)0, c, false, false);
}

// ---------------------------------------------------------------------------
// Kernel 0: one-time fp32 -> bf16 weight conversion (Wq, Wk, Wv, Wo).
// grid (512, 4), block 256; each thread converts 8 elements.
// ---------------------------------------------------------------------------
__global__ __launch_bounds__(256)
void cvt_weights_kernel(const float* __restrict__ s0, const float* __restrict__ s1,
                        const float* __restrict__ s2, const float* __restrict__ s3,
                        __bf16* __restrict__ d0, __bf16* __restrict__ d1,
                        __bf16* __restrict__ d2, __bf16* __restrict__ d3) {
  const int w = blockIdx.y;
  const float* s = w == 0 ? s0 : (w == 1 ? s1 : (w == 2 ? s2 : s3));
  __bf16*      d = w == 0 ? d0 : (w == 1 ? d1 : (w == 2 ? d2 : d3));
  const size_t i = ((size_t)blockIdx.x * 256 + threadIdx.x) * 8;
  const float4 a = *(const float4*)(s + i);
  const float4 b = *(const float4*)(s + i + 4);
  d[i + 0] = (__bf16)a.x; d[i + 1] = (__bf16)a.y;
  d[i + 2] = (__bf16)a.z; d[i + 3] = (__bf16)a.w;
  d[i + 4] = (__bf16)b.x; d[i + 5] = (__bf16)b.y;
  d[i + 6] = (__bf16)b.z; d[i + 7] = (__bf16)b.w;
}

// ---------------------------------------------------------------------------
// Kernel 1: fused QKV projection.  out = X @ W^T + b, bf16 outputs.
// q,k stored (b,h,l,d); v stored transposed (b,h,d,l).
// grid (M/128, N/64, 3), block 128 (4 waves, wave w owns rows 32w..32w+31).
// ---------------------------------------------------------------------------
__global__ __launch_bounds__(128)
void qkv_proj_kernel(const float* __restrict__ Qin, const float* __restrict__ Kin,
                     const float* __restrict__ Vin,
                     const __bf16* __restrict__ Wq, const float* __restrict__ bq,
                     const __bf16* __restrict__ Wk, const float* __restrict__ bk,
                     const __bf16* __restrict__ Wv, const float* __restrict__ bv,
                     __bf16* __restrict__ qo, __bf16* __restrict__ ko,
                     __bf16* __restrict__ vo) {
  __shared__ __bf16 Xs[128][40];   // 80B rows: 16B aligned, conflict-free
  __shared__ __bf16 Ws[64][40];

  const int which = blockIdx.z;
  const float*  X    = which == 0 ? Qin : (which == 1 ? Kin : Vin);
  const __bf16* W    = which == 0 ? Wq  : (which == 1 ? Wk  : Wv);
  const float*  bias = which == 0 ? bq  : (which == 1 ? bk  : bv);
  __bf16*       out  = which == 0 ? qo  : (which == 1 ? ko  : vo);

  const int m0 = blockIdx.x * 128;
  const int n0 = blockIdx.y * 64;
  const int tid = threadIdx.x, wv = tid >> 5, lane = tid & 31;

  const f32x8 zf = {0.f, 0.f, 0.f, 0.f, 0.f, 0.f, 0.f, 0.f};
  f32x8 acc[2][4] = {{zf, zf, zf, zf}, {zf, zf, zf, zf}};

  for (int kk = 0; kk < DMODEL; kk += 32) {
    // W tile (bf16): async copy to LDS, 2 x b128 per thread
#pragma unroll
    for (int it = 0; it < 2; ++it) {
      const int i = tid + it * 128;
      const int r = i >> 2, c = (i & 3) * 8;
      async_copy16(W + (size_t)(n0 + r) * DMODEL + kk + c, &Ws[r][c]);
    }
    // X tile (fp32 -> bf16): 8 x float4 per thread, convert, ds_store
#pragma unroll
    for (int it = 0; it < 8; ++it) {
      const int i = tid + it * 128;
      const int r = i >> 3, c = (i & 7) * 4;
      const float4 xv = *(const float4*)(X + (size_t)(m0 + r) * DMODEL + kk + c);
      Xs[r][c + 0] = (__bf16)xv.x; Xs[r][c + 1] = (__bf16)xv.y;
      Xs[r][c + 2] = (__bf16)xv.z; Xs[r][c + 3] = (__bf16)xv.w;
    }
    wait_async<0>();
    __syncthreads();
#pragma unroll
    for (int a = 0; a < 2; ++a) {
      const bf16x16 af = frag_a(&Xs[wv * 32 + a * 16][0], 40);
#pragma unroll
      for (int j = 0; j < 4; ++j) {
        const bf16x16 bf = frag_bT(&Ws[j * 16][0], 40);
        acc[a][j] = wmma_bf16(af, bf, acc[a][j]);
      }
    }
    __syncthreads();
  }

  // epilogue: bias add, scatter into head-split layout
  const int Mh = (lane >> 4) << 3;
#pragma unroll
  for (int j = 0; j < 4; ++j) {
    const int n = n0 + j * 16 + (lane & 15);
    const float bs = bias[n];
    const int h = n >> 6, d = n & 63;
#pragma unroll
    for (int a = 0; a < 2; ++a) {
#pragma unroll
      for (int r = 0; r < 8; ++r) {
        const int m = m0 + wv * 32 + a * 16 + Mh + r;
        const int b = m >> 11, l = m & (L_ - 1);
        const float v = acc[a][j][r] + bs;
        size_t addr;
        if (which == 2)  // v transposed: (b,h,d,l)
          addr = ((size_t)(b * H_ + h) * DHEAD + d) * L_ + l;
        else             // q,k: (b,h,l,d)
          addr = ((size_t)(b * H_ + h) * L_ + l) * DHEAD + d;
        out[addr] = (__bf16)v;
      }
    }
  }
}

// ---------------------------------------------------------------------------
// Kernel 2: flash attention, double-buffered async K/V staging.
// grid (L/64, H, B), block 128 (4 waves); wave owns 16 query rows.
// ---------------------------------------------------------------------------
__device__ __forceinline__ void stage_kv(const __bf16* kb, const __bf16* vb,
                                         int t, __bf16 (*Ks)[72],
                                         __bf16 (*Vts)[72], int tid) {
#pragma unroll
  for (int it = 0; it < 4; ++it) {      // 4+4 async b128 per thread
    const int i = tid + it * 128;
    const int r = i >> 3, c = (i & 7) * 8;
    async_copy16(kb + (size_t)(t + r) * DHEAD + c, &Ks[r][c]);   // Ks[key][d]
    async_copy16(vb + (size_t)r * L_ + t + c, &Vts[r][c]);       // Vts[d][key]
  }
}

__global__ __launch_bounds__(128)
void attn_kernel(const __bf16* __restrict__ q, const __bf16* __restrict__ k,
                 const __bf16* __restrict__ vT, __bf16* __restrict__ aout) {
  __shared__ __bf16 Ks[2][64][72];
  __shared__ __bf16 Vts[2][64][72];
  __shared__ __bf16 Ps[4][16][72];   // per-wave P tile (C->A layout bounce)

  const int qt = blockIdx.x, h = blockIdx.y, b = blockIdx.z;
  const int bh = b * H_ + h;
  const int q0 = qt * 64;
  const int tid = threadIdx.x, wv = tid >> 5, lane = tid & 31;

  const __bf16* kbase = k + (size_t)bh * L_ * DHEAD;
  const __bf16* vbase = vT + (size_t)bh * DHEAD * L_;

  const __bf16* qbase = q + ((size_t)bh * L_ + q0 + wv * 16) * DHEAD;
  const bf16x16 qf0 = frag_a(qbase, DHEAD);        // d = 0..31
  const bf16x16 qf1 = frag_a(qbase + 32, DHEAD);   // d = 32..63

  const f32x8 zf = {0.f, 0.f, 0.f, 0.f, 0.f, 0.f, 0.f, 0.f};
  f32x8 accO[4] = {zf, zf, zf, zf};
  float mr[8], lr[8];
#pragma unroll
  for (int r = 0; r < 8; ++r) { mr[r] = -1e30f; lr[r] = 0.f; }

  const float scale = 0.125f;  // 1/sqrt(64)

  stage_kv(kbase, vbase, 0, Ks[0], Vts[0], tid);   // prefetch tile 0
  int cur = 0;

  for (int t = 0; t < L_; t += 64) {
    if (t + 64 < L_) {                             // prefetch next tile
      stage_kv(kbase, vbase, t + 64, Ks[cur ^ 1], Vts[cur ^ 1], tid);
      wait_async<8>();                             // current tile's 8 are done
    } else {
      wait_async<0>();
    }
    __syncthreads();

    // S = q . k^T  (16 x 64 per wave)
    f32x8 s[4] = {zf, zf, zf, zf};
#pragma unroll
    for (int j = 0; j < 4; ++j) {
      s[j] = wmma_bf16(qf0, frag_bT(&Ks[cur][j * 16][0], 72), s[j]);
      s[j] = wmma_bf16(qf1, frag_bT(&Ks[cur][j * 16][32], 72), s[j]);
    }

    // online softmax; each row M lives in one lane-half (C/D layout)
#pragma unroll
    for (int r = 0; r < 8; ++r) {
      float sm = fmaxf(fmaxf(s[0][r], s[1][r]), fmaxf(s[2][r], s[3][r])) * scale;
#pragma unroll
      for (int off = 8; off > 0; off >>= 1) sm = fmaxf(sm, __shfl_xor(sm, off, 32));
      const float mnew = fmaxf(mr[r], sm);
      const float corr = __expf(mr[r] - mnew);
      float rs = 0.f;
#pragma unroll
      for (int j = 0; j < 4; ++j) {
        const float p = __expf(s[j][r] * scale - mnew);
        s[j][r] = p; rs += p;
      }
#pragma unroll
      for (int off = 8; off > 0; off >>= 1) rs += __shfl_xor(rs, off, 32);
      mr[r] = mnew;
      lr[r] = lr[r] * corr + rs;
#pragma unroll
      for (int j = 0; j < 4; ++j) accO[j][r] *= corr;
    }

    // P: accumulator C-layout -> A-layout via per-wave LDS bounce
    const int Mh = (lane >> 4) << 3;
#pragma unroll
    for (int j = 0; j < 4; ++j)
#pragma unroll
      for (int r = 0; r < 8; ++r)
        Ps[wv][Mh + r][j * 16 + (lane & 15)] = (__bf16)s[j][r];
    asm volatile("s_wait_dscnt 0" ::: "memory");  // same-wave LDS RAW

    // O += P . V   (V^T rows are B^T rows)
#pragma unroll
    for (int c = 0; c < 2; ++c) {
      const bf16x16 pf = frag_a(&Ps[wv][0][c * 32], 72);
#pragma unroll
      for (int j = 0; j < 4; ++j)
        accO[j] = wmma_bf16(pf, frag_bT(&Vts[cur][j * 16][c * 32], 72), accO[j]);
    }
    __syncthreads();
    cur ^= 1;
  }

  // epilogue: O / l, store bf16 into (b, l, h*64+d) for the final GEMM
  const int Mh = (lane >> 4) << 3;
  float invl[8];
#pragma unroll
  for (int r = 0; r < 8; ++r) invl[r] = 1.f / lr[r];
#pragma unroll
  for (int j = 0; j < 4; ++j) {
    const int d = j * 16 + (lane & 15);
#pragma unroll
    for (int r = 0; r < 8; ++r) {
      const int lq = q0 + wv * 16 + Mh + r;
      aout[((size_t)(b * L_ + lq)) * (H_ * DHEAD) + h * DHEAD + d] =
          (__bf16)(accO[j][r] * invl[r]);
    }
  }
}

// ---------------------------------------------------------------------------
// Kernel 3: output projection.  out = A @ Wo^T + bo, fp32 result.
// grid (M/128, N/64), block 128.  Both tiles staged async (bf16).
// ---------------------------------------------------------------------------
__global__ __launch_bounds__(128)
void out_proj_kernel(const __bf16* __restrict__ A, const __bf16* __restrict__ Wo,
                     const float* __restrict__ bo, float* __restrict__ out) {
  __shared__ __bf16 Xs[128][40];
  __shared__ __bf16 Ws[64][40];

  const int m0 = blockIdx.x * 128;
  const int n0 = blockIdx.y * 64;
  const int tid = threadIdx.x, wv = tid >> 5, lane = tid & 31;

  const f32x8 zf = {0.f, 0.f, 0.f, 0.f, 0.f, 0.f, 0.f, 0.f};
  f32x8 acc[2][4] = {{zf, zf, zf, zf}, {zf, zf, zf, zf}};

  for (int kk = 0; kk < DMODEL; kk += 32) {
#pragma unroll
    for (int it = 0; it < 4; ++it) {          // A tile: 4 x b128 per thread
      const int i = tid + it * 128;
      const int r = i >> 2, c = (i & 3) * 8;
      async_copy16(A + (size_t)(m0 + r) * DMODEL + kk + c, &Xs[r][c]);
    }
#pragma unroll
    for (int it = 0; it < 2; ++it) {          // W tile: 2 x b128 per thread
      const int i = tid + it * 128;
      const int r = i >> 2, c = (i & 3) * 8;
      async_copy16(Wo + (size_t)(n0 + r) * DMODEL + kk + c, &Ws[r][c]);
    }
    wait_async<0>();
    __syncthreads();
#pragma unroll
    for (int a = 0; a < 2; ++a) {
      const bf16x16 af = frag_a(&Xs[wv * 32 + a * 16][0], 40);
#pragma unroll
      for (int j = 0; j < 4; ++j)
        acc[a][j] = wmma_bf16(af, frag_bT(&Ws[j * 16][0], 40), acc[a][j]);
    }
    __syncthreads();
  }

  const int Mh = (lane >> 4) << 3;
#pragma unroll
  for (int j = 0; j < 4; ++j) {
    const int n = n0 + j * 16 + (lane & 15);
    const float bs = bo[n];
#pragma unroll
    for (int a = 0; a < 2; ++a)
#pragma unroll
      for (int r = 0; r < 8; ++r)
        out[(size_t)(m0 + wv * 32 + a * 16 + Mh + r) * DMODEL + n] =
            acc[a][j][r] + bs;
  }
}

// ---------------------------------------------------------------------------
extern "C" void kernel_launch(void* const* d_in, const int* in_sizes, int n_in,
                              void* d_out, int out_size, void* d_ws, size_t ws_size,
                              hipStream_t stream) {
  const float* V  = (const float*)d_in[0];
  const float* K  = (const float*)d_in[1];
  const float* Q  = (const float*)d_in[2];
  const float* Wv = (const float*)d_in[3];
  const float* bv = (const float*)d_in[4];
  const float* Wk = (const float*)d_in[5];
  const float* bk = (const float*)d_in[6];
  const float* Wq = (const float*)d_in[7];
  const float* bq = (const float*)d_in[8];
  const float* Wo = (const float*)d_in[9];
  const float* bo = (const float*)d_in[10];
  float* out = (float*)d_out;

  const size_t wel = (size_t)DMODEL * DMODEL;          // 1,048,576
  const size_t ael = (size_t)B_ * H_ * L_ * DHEAD;     // 8,388,608
  __bf16* wqb = (__bf16*)d_ws;
  __bf16* wkb = wqb + wel;
  __bf16* wvb = wkb + wel;
  __bf16* wob = wvb + wel;
  __bf16* qb  = wob + wel;
  __bf16* kb  = qb + ael;
  __bf16* vT  = kb + ael;
  __bf16* ab  = vT + ael;                              // 72 MB total

  cvt_weights_kernel<<<dim3(512, 4), 256, 0, stream>>>(
      Wq, Wk, Wv, Wo, wqb, wkb, wvb, wob);
  qkv_proj_kernel<<<dim3(64, 16, 3), 128, 0, stream>>>(
      Q, K, V, wqb, bq, wkb, bk, wvb, bv, qb, kb, vT);
  attn_kernel<<<dim3(L_ / 64, H_, B_), 128, 0, stream>>>(qb, kb, vT, ab);
  out_proj_kernel<<<dim3(64, 16), 128, 0, stream>>>(ab, wob, bo, out);
}
